// RPN_26379689132801
// MI455X (gfx1250) — matrix operand order repaired and checked by reading, not compiled, not run on hardware
//
#include <hip/hip_runtime.h>
#include <hip/hip_bf16.h>
#include <math.h>

#define IN_CHN   256
#define NUM_AN   9
#define FEAT_HW  200
#define NPIX     40000
#define KDIM     2304
#define PADW     202
#define PADSZ    (PADW*PADW)
#define IMG_SZ   1600.0f
#define PRENMS   2000
#define TOPK_N   1000
#define NMS_THR  0.7f
#define MIN_SZ   16.0f
#define BBOXCLIP 4.135166556742356f   /* log(1000/16) */
#define NCAND    2048

// conv LDS layout (dynamic shared, base offset 0): double-buffered A(8KB)+B(4KB)
#define LDS_B_OFF 8192
#define LDS_BUFSZ 12288
#define LDS_TOTAL (2 * LDS_BUFSZ)

typedef __attribute__((ext_vector_type(16))) __bf16 v16bf;
typedef __attribute__((ext_vector_type(8)))  float  v8f;

union Frag { v16bf v; uint4 q[2]; };

__device__ __forceinline__ unsigned short f2bf(float f) {
  unsigned u = __float_as_uint(f);
  u += 0x7fffu + ((u >> 16) & 1u);   // round-to-nearest-even
  return (unsigned short)(u >> 16);
}

// ---------------- prep kernels ----------------
__global__ void pad_feat_k(const float* __restrict__ feat, unsigned short* __restrict__ fp) {
  int idx = blockIdx.x * 256 + threadIdx.x;
  if (idx >= IN_CHN * PADSZ) return;
  int ic = idx / PADSZ;
  int r  = idx - ic * PADSZ;
  int ph = r / PADW, pw = r - ph * PADW;
  int h = ph - 1, w = pw - 1;
  float v = 0.f;
  if ((unsigned)h < FEAT_HW && (unsigned)w < FEAT_HW)
    v = feat[ic * NPIX + h * FEAT_HW + w];
  fp[idx] = f2bf(v);
}

__global__ void cvt_w_k(const float* __restrict__ w, unsigned short* __restrict__ o, int n) {
  int i = blockIdx.x * 256 + threadIdx.x;
  if (i < n) o[i] = f2bf(w[i]);
}

__global__ void build_w45_k(const float* __restrict__ cls_w, const float* __restrict__ reg_w,
                            unsigned short* __restrict__ w45) {
  int i = blockIdx.x * 256 + threadIdx.x;
  if (i >= 48 * IN_CHN) return;
  int j = i >> 8, c = i & 255;
  float v = 0.f;
  if (j < 9)        v = cls_w[j * IN_CHN + c];
  else if (j < 45)  v = reg_w[(j - 9) * IN_CHN + c];
  w45[i] = f2bf(v);
}

__global__ void zero_k(int* p, int n) {
  int i = blockIdx.x * 256 + threadIdx.x;
  if (i < n) p[i] = 0;
}

// ---------------- conv 3x3 as implicit GEMM (WMMA bf16) ----------------
// M=256 (oc), N=40000 (pixel), K=2304. Tile 128x64, BK=32, 8 waves.
// A tile streamed with GLOBAL_LOAD_ASYNC_TO_LDS_B128 (ASYNCcnt), double-buffered LDS.
__global__ __launch_bounds__(256) void conv_gemm_k(
    const unsigned short* __restrict__ fp,    // padded feat bf16 [256][202][202]
    const unsigned short* __restrict__ wbf,   // weights bf16 [256][2304]
    const float* __restrict__ conv_b,
    unsigned short* __restrict__ act)         // out: bf16 [pixel][256]
{
  extern __shared__ char smem[];              // dynamic LDS => base byte offset 0
  const int tid  = threadIdx.x;
  const int lane = tid & 31, wid = tid >> 5;
  const int wm = wid & 3, wn = wid >> 2;
  const int n_block = blockIdx.x * 64;
  const int m_block = blockIdx.y * 128;

  v8f acc[2][2] = {};

  const int a_row = tid >> 1;
  const int a_kc  = (tid & 1) * 16;
  const int b_n   = tid >> 2;
  const int b_kq  = (tid & 3) * 8;
  const int pix   = n_block + b_n;
  const int ph0   = pix / FEAT_HW;
  const int pw0   = pix - ph0 * FEAT_HW;
  const int lo    = lane & 15;
  const int hi    = lane >> 4;

  // async A-tile load: each thread moves 32B straight into LDS (no VGPR staging)
  const unsigned lds_a_base = (unsigned)(a_row * 64 + a_kc * 2);
  auto load_a_async = [&](int k0, int buf) {
    unsigned lds = (unsigned)(buf * LDS_BUFSZ) + lds_a_base;
    unsigned long long ga =
        (unsigned long long)(uintptr_t)(&wbf[(m_block + a_row) * KDIM + k0 + a_kc]);
    asm volatile(
        "global_load_async_to_lds_b128 %0, %1, off\n\t"
        "global_load_async_to_lds_b128 %0, %1, off offset:16"
        :: "v"(lds), "v"(ga) : "memory");
  };
  // B tile: gather 64 pixels x 32 k from the padded feature map into LDS
  auto load_b = [&](int k0, int buf) {
    unsigned short* Bs = (unsigned short*)(smem + buf * LDS_BUFSZ + LDS_B_OFF);
#pragma unroll
    for (int kk = 0; kk < 8; ++kk) {
      int kg = k0 + b_kq + kk;
      int ic = kg / 9;
      int rr = kg - ic * 9;
      int kh = rr / 3;
      int kw = rr - kh * 3;
      Bs[b_n * 32 + b_kq + kk] = fp[ic * PADSZ + (ph0 + kh) * PADW + (pw0 + kw)];
    }
  };

  load_a_async(0, 0);
  load_b(0, 0);

  int buf = 0;
  for (int k0 = 0; k0 < KDIM; k0 += 32) {
    asm volatile("s_wait_asynccnt 0x0" ::: "memory");  // own wave's async A-load done
    __syncthreads();                                   // all waves' tiles visible
    if (k0 + 32 < KDIM) {                              // prefetch next tile
      load_a_async(k0 + 32, buf ^ 1);
      load_b(k0 + 32, buf ^ 1);
    }
    const unsigned short* As = (const unsigned short*)(smem + buf * LDS_BUFSZ);
    const unsigned short* Bs = (const unsigned short*)(smem + buf * LDS_BUFSZ + LDS_B_OFF);
#pragma unroll
    for (int tm = 0; tm < 2; ++tm) {
      int m = wm * 32 + tm * 16 + lo;
      Frag a;
      a.q[0] = *reinterpret_cast<const uint4*>(&As[m * 32 + hi * 8]);
      a.q[1] = *reinterpret_cast<const uint4*>(&As[m * 32 + hi * 8 + 16]);
#pragma unroll
      for (int tn = 0; tn < 2; ++tn) {
        int n = wn * 32 + tn * 16 + lo;
        Frag b;
        const uint4* bp = reinterpret_cast<const uint4*>(&Bs[n * 32 + hi * 8]);
        b.q[0] = bp[0];
        b.q[1] = bp[2];
        acc[tm][tn] = __builtin_amdgcn_wmma_f32_16x16x32_bf16(
            false, a.v, false, b.v, (short)0, acc[tm][tn], false, false);
      }
    }
    buf ^= 1;
  }

  // epilogue: bias + ReLU, pack to bf16, store pixel-major [n][oc]
#pragma unroll
  for (int tm = 0; tm < 2; ++tm) {
    int m_base = m_block + wm * 32 + tm * 16 + hi * 8;
#pragma unroll
    for (int tn = 0; tn < 2; ++tn) {
      int n = n_block + wn * 32 + tn * 16 + lo;
      unsigned pk[4];
#pragma unroll
      for (int r = 0; r < 4; ++r) {
        float v0 = fmaxf(acc[tm][tn][2 * r]     + conv_b[m_base + 2 * r],     0.f);
        float v1 = fmaxf(acc[tm][tn][2 * r + 1] + conv_b[m_base + 2 * r + 1], 0.f);
        pk[r] = (unsigned)f2bf(v0) | ((unsigned)f2bf(v1) << 16);
      }
      *reinterpret_cast<uint4*>(&act[n * IN_CHN + m_base]) = make_uint4(pk[0], pk[1], pk[2], pk[3]);
    }
  }
}

// ---------------- head GEMM (cls+reg fused): D[pixel][48] = act[pixel][c] * w45[j][c]^T ----
__global__ __launch_bounds__(128) void head_gemm_k(
    const unsigned short* __restrict__ act,
    const unsigned short* __restrict__ w45,
    float* __restrict__ out45)
{
  const int tid  = threadIdx.x;
  const int lane = tid & 31, wid = tid >> 5;
  const int p0 = (blockIdx.x * 4 + wid) * 16;
  const int lo = lane & 15, hi = lane >> 4;
  v8f acc[3] = {};
  for (int k0 = 0; k0 < IN_CHN; k0 += 32) {
    Frag a;  // A = 16 pixels x 32 channels
    const unsigned short* ap = &act[(p0 + lo) * IN_CHN + k0 + hi * 8];
    a.q[0] = *reinterpret_cast<const uint4*>(ap);
    a.q[1] = *reinterpret_cast<const uint4*>(ap + 16);
#pragma unroll
    for (int t = 0; t < 3; ++t) {
      Frag b;  // B = 32 channels x 16 heads
      const unsigned short* bp = &w45[(t * 16 + lo) * IN_CHN + k0 + hi * 8];
      b.q[0] = *reinterpret_cast<const uint4*>(bp);
      b.q[1] = *reinterpret_cast<const uint4*>(bp + 16);
      acc[t] = __builtin_amdgcn_wmma_f32_16x16x32_bf16(
          false, a.v, false, b.v, (short)0, acc[t], false, false);
    }
  }
#pragma unroll
  for (int t = 0; t < 3; ++t) {
    int j = t * 16 + lo;
#pragma unroll
    for (int r = 0; r < 8; ++r) {
      int pixel = p0 + hi * 8 + r;
      out45[pixel * 48 + j] = acc[t][r];
    }
  }
}

// ---------------- anchor decode + sigmoid + histogram ----------------
__global__ void decode_k(const float* __restrict__ out45,
                         const float* __restrict__ cls_b,
                         const float* __restrict__ reg_b,
                         float* __restrict__ scores,
                         float* __restrict__ props,
                         int* __restrict__ hist1)
{
  int i = blockIdx.x * 256 + threadIdx.x;
  if (i >= NPIX * NUM_AN) return;
  int pixel = i / NUM_AN;
  int a = i - pixel * NUM_AN;
  const float* row = &out45[pixel * 48];
  float logit = row[a] + cls_b[a];
  float s = 1.f / (1.f + __expf(-logit));
  float dx = row[9 + a * 4 + 0] + reg_b[a * 4 + 0];
  float dy = row[9 + a * 4 + 1] + reg_b[a * 4 + 1];
  float dw = fminf(row[9 + a * 4 + 2] + reg_b[a * 4 + 2], BBOXCLIP);
  float dh = fminf(row[9 + a * 4 + 3] + reg_b[a * 4 + 3], BBOXCLIP);
  // anchor (a = ar_idx*3 + scale_idx)
  int ar_i = a / 3, s_i = a - ar_i * 3;
  const float ars[3] = {0.5f, 1.0f, 2.0f};
  const float scs[3] = {128.f, 256.f, 512.f};
  float hr = sqrtf(ars[ar_i]);
  float wr = 1.f / hr;
  float wsz = wr * scs[s_i], hsz = hr * scs[s_i];
  float bx0 = rintf(-wsz * 0.5f), by0 = rintf(-hsz * 0.5f);
  float bx1 = rintf( wsz * 0.5f), by1 = rintf( hsz * 0.5f);
  int gy = pixel / FEAT_HW, gx = pixel - gy * FEAT_HW;
  float sx = gx * 8.f, sy = gy * 8.f;
  float ax0 = sx + bx0, ay0 = sy + by0, ax1 = sx + bx1, ay1 = sy + by1;
  float w  = ax1 - ax0, h = ay1 - ay0;
  float cx = ax0 + 0.5f * w, cy = ay0 + 0.5f * h;
  float pcx = dx * w + cx, pcy = dy * h + cy;
  float pw = __expf(dw) * w, ph = __expf(dh) * h;
  props[i * 4 + 0] = pcx - 0.5f * pw;
  props[i * 4 + 1] = pcy - 0.5f * ph;
  props[i * 4 + 2] = pcx + 0.5f * pw;
  props[i * 4 + 3] = pcy + 0.5f * ph;
  scores[i] = s;
  atomicAdd(&hist1[__float_as_uint(s) >> 24], 1);
}

// ---------------- radix-threshold top-2000 ----------------
__global__ void thresh1_k(const int* __restrict__ hist1, int* __restrict__ sel) {
  if (threadIdx.x != 0) return;
  int cum = 0;
  for (int b = 255; b >= 0; --b) {
    int c = hist1[b];
    if (cum + c >= PRENMS || b == 0) { sel[0] = b; sel[1] = cum; return; }
    cum += c;
  }
}

__global__ void hist2_k(const float* __restrict__ scores, const int* __restrict__ sel,
                        int* __restrict__ hist2) {
  int i = blockIdx.x * 256 + threadIdx.x;
  if (i >= NPIX * NUM_AN) return;
  unsigned u = __float_as_uint(scores[i]);
  if ((int)(u >> 24) == sel[0]) atomicAdd(&hist2[(u >> 16) & 255], 1);
}

__global__ void thresh2_k(const int* __restrict__ hist2, int* __restrict__ sel) {
  if (threadIdx.x != 0) return;
  int target = PRENMS - sel[1];
  int cum = 0, t2 = 0;
  for (int b = 255; b >= 0; --b) {
    int c = hist2[b];
    if (cum + c >= target || b == 0) { t2 = b; break; }
    cum += c;
  }
  sel[2] = (sel[0] << 8) | t2;
}

__global__ void compact_k(const float* __restrict__ scores, const float* __restrict__ props,
                          int* __restrict__ sel, float* __restrict__ cs, float* __restrict__ cb) {
  int i = blockIdx.x * 256 + threadIdx.x;
  if (i >= NPIX * NUM_AN) return;
  float s = scores[i];
  unsigned u = __float_as_uint(s);
  if ((int)(u >> 16) < sel[2]) return;
  int pos = atomicAdd(&sel[3], 1);
  if (pos >= NCAND) return;
  float x0 = fminf(fmaxf(props[i * 4 + 0], 0.f), IMG_SZ);
  float y0 = fminf(fmaxf(props[i * 4 + 1], 0.f), IMG_SZ);
  float x1 = fminf(fmaxf(props[i * 4 + 2], 0.f), IMG_SZ);
  float y1 = fminf(fmaxf(props[i * 4 + 3], 0.f), IMG_SZ);
  bool valid = (x1 - x0 >= MIN_SZ) && (y1 - y0 >= MIN_SZ);
  cs[pos] = valid ? s : -1.f;
  cb[pos * 4 + 0] = x0; cb[pos * 4 + 1] = y0;
  cb[pos * 4 + 2] = x1; cb[pos * 4 + 3] = y1;
}

// ---------------- single-block sort + NMS + output ----------------
__device__ void bitonic_desc(float* ss, float* x0, float* y0, float* x1, float* y1, int tid) {
  for (int k2 = 2; k2 <= NCAND; k2 <<= 1) {
    for (int j = k2 >> 1; j > 0; j >>= 1) {
      for (int t = tid; t < NCAND; t += 1024) {
        int x = t ^ j;
        if (x > t) {
          bool up = ((t & k2) == 0);
          float a = ss[t], b = ss[x];
          bool sw = up ? (a < b) : (a > b);
          if (sw) {
            ss[t] = b; ss[x] = a;
            float tv;
            tv = x0[t]; x0[t] = x0[x]; x0[x] = tv;
            tv = y0[t]; y0[t] = y0[x]; y0[x] = tv;
            tv = x1[t]; x1[t] = x1[x]; x1[x] = tv;
            tv = y1[t]; y1[t] = y1[x]; y1[x] = tv;
          }
        }
      }
      __syncthreads();
    }
  }
}

__global__ __launch_bounds__(1024) void nms_final_k(const int* __restrict__ sel,
                                                    const float* __restrict__ cs,
                                                    const float* __restrict__ cb,
                                                    float* __restrict__ out) {
  __shared__ float ss[NCAND], bx0[NCAND], by0[NCAND], bx1[NCAND], by1[NCAND];
  __shared__ unsigned char keep[NCAND];
  const int tid = threadIdx.x;
  int cnt = sel[3];
  if (cnt > NCAND) cnt = NCAND;
  for (int t = tid; t < NCAND; t += 1024) {
    if (t < cnt) {
      ss[t] = cs[t];
      bx0[t] = cb[t * 4 + 0]; by0[t] = cb[t * 4 + 1];
      bx1[t] = cb[t * 4 + 2]; by1[t] = cb[t * 4 + 3];
    } else {
      ss[t] = -1.f; bx0[t] = by0[t] = bx1[t] = by1[t] = 0.f;
    }
  }
  __syncthreads();
  bitonic_desc(ss, bx0, by0, bx1, by1, tid);
  for (int t = tid; t < NCAND; t += 1024) {
    if (t >= PRENMS) ss[t] = -1.f;
    keep[t] = (ss[t] > 0.f) ? 1 : 0;
  }
  __syncthreads();
  // greedy NMS over first 2000 (uniform branch on keep[i])
  for (int i = 0; i < PRENMS; ++i) {
    if (keep[i]) {
      float ax0 = bx0[i], ay0 = by0[i], ax1 = bx1[i], ay1 = by1[i];
      float areai = (ax1 - ax0) * (ay1 - ay0);
      for (int j = i + 1 + tid; j < PRENMS; j += 1024) {
        if (!keep[j]) continue;
        float ix0 = fmaxf(ax0, bx0[j]), iy0 = fmaxf(ay0, by0[j]);
        float ix1 = fminf(ax1, bx1[j]), iy1 = fminf(ay1, by1[j]);
        float iw = fmaxf(ix1 - ix0, 0.f), ih = fmaxf(iy1 - iy0, 0.f);
        float inter = iw * ih;
        float areaj = (bx1[j] - bx0[j]) * (by1[j] - by0[j]);
        float iou = inter / (areai + areaj - inter + 1e-9f);
        if (iou > NMS_THR) keep[j] = 0;
      }
    }
    __syncthreads();
  }
  for (int t = tid; t < NCAND; t += 1024)
    if (!keep[t]) ss[t] = -1.f;
  __syncthreads();
  bitonic_desc(ss, bx0, by0, bx1, by1, tid);
  for (int t = tid; t < TOPK_N; t += 1024) {
    float s = ss[t];
    bool pos = s > 0.f;
    out[t * 4 + 0] = pos ? bx0[t] : 0.f;
    out[t * 4 + 1] = pos ? by0[t] : 0.f;
    out[t * 4 + 2] = pos ? bx1[t] : 0.f;
    out[t * 4 + 3] = pos ? by1[t] : 0.f;
    out[4 * TOPK_N + t] = pos ? s : 0.f;
  }
}

extern "C" void kernel_launch(void* const* d_in, const int* in_sizes, int n_in,
                              void* d_out, int out_size, void* d_ws, size_t ws_size,
                              hipStream_t stream) {
  (void)in_sizes; (void)n_in; (void)out_size; (void)ws_size;
  const float* feat   = (const float*)d_in[1];
  const float* conv_w = (const float*)d_in[2];
  const float* conv_b = (const float*)d_in[3];
  const float* cls_w  = (const float*)d_in[4];
  const float* cls_b  = (const float*)d_in[5];
  const float* reg_w  = (const float*)d_in[6];
  const float* reg_b  = (const float*)d_in[7];
  float* out = (float*)d_out;

  char* ws = (char*)d_ws;
  size_t off = 0;
  auto take = [&](size_t bytes) -> void* {
    void* p = ws + off;
    off += (bytes + 255) & ~(size_t)255;
    return p;
  };
  unsigned short* feat_pad = (unsigned short*)take((size_t)IN_CHN * PADSZ * 2);
  unsigned short* wbf      = (unsigned short*)take((size_t)IN_CHN * KDIM * 2);
  unsigned short* w45      = (unsigned short*)take((size_t)48 * IN_CHN * 2);
  unsigned short* act      = (unsigned short*)take((size_t)NPIX * IN_CHN * 2);
  float*          out45    = (float*)take((size_t)NPIX * 48 * 4);
  float*          scores   = (float*)take((size_t)NPIX * NUM_AN * 4);
  float*          props    = (float*)take((size_t)NPIX * NUM_AN * 16);
  int*            hist1    = (int*)take(256 * 4);
  int*            hist2    = (int*)take(256 * 4);
  int*            sel      = (int*)take(8 * 4);
  float*          cs       = (float*)take((size_t)NCAND * 4);
  float*          cb       = (float*)take((size_t)NCAND * 16);

  const int nAll = NPIX * NUM_AN;           // 360000
  const int gAll = (nAll + 255) / 256;      // 1407

  // prep
  pad_feat_k<<<(IN_CHN * PADSZ) / 256, 256, 0, stream>>>(feat, feat_pad);
  cvt_w_k<<<(IN_CHN * KDIM) / 256, 256, 0, stream>>>(conv_w, wbf, IN_CHN * KDIM);
  build_w45_k<<<(48 * IN_CHN) / 256, 256, 0, stream>>>(cls_w, reg_w, w45);
  zero_k<<<1, 256, 0, stream>>>(hist1, 256);
  zero_k<<<1, 256, 0, stream>>>(hist2, 256);
  zero_k<<<1, 256, 0, stream>>>(sel, 8);

  // conv 3x3 + ReLU  (WMMA + async global->LDS, double-buffered)
  conv_gemm_k<<<dim3(NPIX / 64, IN_CHN / 128), 256, LDS_TOTAL, stream>>>(feat_pad, wbf, conv_b, act);
  // cls + reg heads  (WMMA)
  head_gemm_k<<<NPIX / 64, 128, 0, stream>>>(act, w45, out45);
  // decode + score histogram
  decode_k<<<gAll, 256, 0, stream>>>(out45, cls_b, reg_b, scores, props, hist1);
  // top-2000 threshold
  thresh1_k<<<1, 64, 0, stream>>>(hist1, sel);
  hist2_k<<<gAll, 256, 0, stream>>>(scores, sel, hist2);
  thresh2_k<<<1, 64, 0, stream>>>(hist2, sel);
  compact_k<<<gAll, 256, 0, stream>>>(scores, props, sel, cs, cb);
  // sort + NMS + final top-1000
  nms_final_k<<<1, 1024, 0, stream>>>(sel, cs, cb, out);
}
